// MHAEncoderFusedProj_44530220924956
// MI455X (gfx1250) — matrix operand order repaired and checked by reading, not compiled, not run on hardware
//
#include <hip/hip_runtime.h>
#include <cstddef>

// ---------------------------------------------------------------------------
// MHA encoder (QKV proj + RoPE + attention + out proj) for gfx1250 (MI455X)
// All matmuls in bf16 via v_wmma_f32_16x16x32_bf16, f32 accumulation.
// Tile staging uses GLOBAL_LOAD_ASYNC_TO_LDS (ASYNCcnt) when available.
// ---------------------------------------------------------------------------

constexpr int Bc   = 4;
constexpr int Sc   = 2048;
constexpr int Ec   = 1024;
constexpr int Hc   = 16;
constexpr int Dc   = 64;
constexpr int Mtot = Bc * Sc;   // 8192 rows
constexpr int Kdim = Ec;        // 1024 reduction dim for projections

typedef __attribute__((ext_vector_type(16))) __bf16       v16bf;
typedef __attribute__((ext_vector_type(8)))  float        v8f;
typedef __attribute__((ext_vector_type(4)))  unsigned int u32x4;
typedef __attribute__((ext_vector_type(4)))  float        f32x4;

union FragAB {            // 16 bf16 = 32B = one WMMA A or B operand
    v16bf v;
    u32x4 q[2];
};

static __device__ __forceinline__ unsigned short f32_to_bf16(float f) {
    unsigned int u = __float_as_uint(f);
    u = (u + 0x7FFFu + ((u >> 16) & 1u)) >> 16;
    return (unsigned short)u;
}
static __device__ __forceinline__ float bf16_to_f32(unsigned short h) {
    return __uint_as_float(((unsigned int)h) << 16);
}

// ---------------------------------------------------------------------------
// Async global->LDS copy (16B per lane), guarded for toolchain support.
// Builtin signature (from hipcc diagnostic): pointee is gcc-vector int4,
// global side AS(1), LDS side AS(3).
// ---------------------------------------------------------------------------
#if defined(__has_builtin)
#  if __has_builtin(__builtin_amdgcn_global_load_async_to_lds_b128)
#    define HAS_ASYNC_LDS 1
#  endif
#endif
#ifndef HAS_ASYNC_LDS
#  define HAS_ASYNC_LDS 0
#endif

typedef int v4i_g __attribute__((vector_size(16)));
typedef __attribute__((address_space(1))) v4i_g* g128p;
typedef __attribute__((address_space(3))) v4i_g* l128p;

static __device__ __forceinline__ void async_copy16(const void* g, void* l) {
#if HAS_ASYNC_LDS
    __builtin_amdgcn_global_load_async_to_lds_b128(
        (g128p)(void*)g, (l128p)l, 0, 0);
#else
    *(u32x4*)l = *(const u32x4*)g;
#endif
}
static __device__ __forceinline__ void async_wait0() {
#if HAS_ASYNC_LDS
#  if __has_builtin(__builtin_amdgcn_s_wait_asynccnt)
    __builtin_amdgcn_s_wait_asynccnt(0);
#  else
    asm volatile("s_wait_asynccnt 0x0" ::: "memory");
#  endif
#endif
}

// ---------------------------------------------------------------------------
// f32 -> bf16 conversion (4 elements / thread)
// ---------------------------------------------------------------------------
__global__ __launch_bounds__(256)
void cvt_f32_bf16(const float* __restrict__ src, unsigned short* __restrict__ dst, int n) {
    int i = (blockIdx.x * 256 + threadIdx.x) * 4;
    if (i < n) {
        f32x4 f = *(const f32x4*)(src + i);
        unsigned int lo = (unsigned int)f32_to_bf16(f.x) | ((unsigned int)f32_to_bf16(f.y) << 16);
        unsigned int hi = (unsigned int)f32_to_bf16(f.z) | ((unsigned int)f32_to_bf16(f.w) << 16);
        *(unsigned int*)(dst + i)     = lo;
        *(unsigned int*)(dst + i + 2) = hi;
    }
}

// ---------------------------------------------------------------------------
// Tiled bf16 WMMA GEMM:  C[M,N] = A[M,K] * Bw[N,K]^T
//   MODE 0: write f32 to outF
//   MODE 1: scatter bf16 into q/k/v buffers laid out (B,H,S,D)
// Block tile 128x128, 8 waves (2x4), wave tile 64x32 (4x2 WMMA accumulators).
// Double-buffered async global->LDS staging overlapped with WMMA compute.
// ---------------------------------------------------------------------------
template<int MODE, int N>
__global__ __launch_bounds__(256)
void gemm_bf16(const unsigned short* __restrict__ A,
               const unsigned short* __restrict__ Bw,
               float* __restrict__ outF,
               unsigned short* __restrict__ qb,
               unsigned short* __restrict__ kb,
               unsigned short* __restrict__ vb)
{
    constexpr int LDA = 40;                    // padded ushort stride (80B, 16B aligned)
    __shared__ unsigned short sA[2][128 * LDA];
    __shared__ unsigned short sB[2][128 * LDA];

    const int tid  = threadIdx.x;
    const int lane = tid & 31;
    const int wave = tid >> 5;
    const int ln   = lane & 15;
    const int lg   = lane >> 4;

    const int n0 = blockIdx.x * 128;
    const int m0 = blockIdx.y * 128;
    const int wm = (wave >> 2) * 64;           // 0 or 64
    const int wn = (wave & 3) * 32;            // 0,32,64,96

    v8f acc[4][2];
    #pragma unroll
    for (int i = 0; i < 4; ++i)
        #pragma unroll
        for (int j = 0; j < 2; ++j)
            #pragma unroll
            for (int r = 0; r < 8; ++r)
                acc[i][j][r] = 0.0f;

    // stage one 128x32 K-slice of A and Bw into LDS buffer `buf`
    auto stage = [&](int k0, int buf) {
        #pragma unroll
        for (int s = 0; s < 2; ++s) {
            int seg = tid + s * 256;           // 512 16B-segments per matrix
            int row = seg >> 2;
            int so  = (seg & 3) * 8;
            async_copy16(&A [(size_t)(m0 + row) * Kdim + k0 + so], &sA[buf][row * LDA + so]);
            async_copy16(&Bw[(size_t)(n0 + row) * Kdim + k0 + so], &sB[buf][row * LDA + so]);
        }
    };

    stage(0, 0);
    async_wait0();
    __syncthreads();

    int cur = 0;
    for (int k0 = 0; k0 < Kdim; k0 += 32) {
        if (k0 + 32 < Kdim) stage(k0 + 32, cur ^ 1);   // prefetch next tile (async)

        FragAB af[4];
        #pragma unroll
        for (int i = 0; i < 4; ++i) {
            const unsigned short* ar = &sA[cur][(wm + i * 16 + ln) * LDA];
            af[i].q[0] = *(const u32x4*)(ar + lg * 8);
            af[i].q[1] = *(const u32x4*)(ar + 16 + lg * 8);
        }
        FragAB bfr[2];
        #pragma unroll
        for (int j = 0; j < 2; ++j) {
            const unsigned short* br = &sB[cur][(wn + j * 16 + ln) * LDA];
            bfr[j].q[0] = *(const u32x4*)(br + lg * 8);
            bfr[j].q[1] = *(const u32x4*)(br + 16 + lg * 8);
        }
        #pragma unroll
        for (int i = 0; i < 4; ++i)
            #pragma unroll
            for (int j = 0; j < 2; ++j)
                acc[i][j] = __builtin_amdgcn_wmma_f32_16x16x32_bf16(
                    false, af[i].v, false, bfr[j].v, (short)0, acc[i][j], false, false);

        async_wait0();
        __syncthreads();
        cur ^= 1;
    }

    // epilogue: C layout = VGPR r, lanes 0-15 -> row r, lanes 16-31 -> row r+8
    #pragma unroll
    for (int i = 0; i < 4; ++i) {
        #pragma unroll
        for (int j = 0; j < 2; ++j) {
            #pragma unroll
            for (int r = 0; r < 8; ++r) {
                int grow = m0 + wm + i * 16 + r + 8 * lg;
                int gcol = n0 + wn + j * 16 + ln;
                float val = acc[i][j][r];
                if constexpr (MODE == 0) {
                    outF[(size_t)grow * N + gcol] = val;
                } else {
                    int which = gcol >> 10;          // 0=q, 1=k, 2=v
                    int rem   = gcol & 1023;
                    int h     = rem >> 6;
                    int d     = rem & 63;
                    int b     = grow >> 11;          // / Sc
                    int s     = grow & (Sc - 1);
                    unsigned short* dst = (which == 0) ? qb : (which == 1) ? kb : vb;
                    dst[(((size_t)(b * Hc + h)) * Sc + s) * Dc + d] = f32_to_bf16(val);
                }
            }
        }
    }
}

// ---------------------------------------------------------------------------
// In-place RoPE over a (B,H,S,D) bf16 buffer. One thread per (row, d<32) pair.
// ---------------------------------------------------------------------------
__global__ __launch_bounds__(256)
void rope_kernel(unsigned short* __restrict__ t,
                 const float* __restrict__ cosb,
                 const float* __restrict__ sinb)
{
    int i   = blockIdx.x * 256 + threadIdx.x;   // over (B*H*S) * 32
    int d   = i & 31;
    int row = i >> 5;                           // (b*H + h)*S + s
    int s   = row & (Sc - 1);
    size_t base = (size_t)row * Dc;
    float t1 = bf16_to_f32(t[base + d]);
    float t2 = bf16_to_f32(t[base + d + 32]);
    float c1 = cosb[s * Dc + d],      s1 = sinb[s * Dc + d];
    float c2 = cosb[s * Dc + d + 32], s2 = sinb[s * Dc + d + 32];
    t[base + d]      = f32_to_bf16(t1 * c1 - t2 * s1);   // rotate_half: -t2 for d<32
    t[base + d + 32] = f32_to_bf16(t2 * c2 + t1 * s2);   // +t1 for d>=32
}

// ---------------------------------------------------------------------------
// Flash attention: grid (S/64, B*H), 128 threads = 4 waves x 16 query rows.
// K tile staged async to LDS; V tile register-transposed; P via per-wave LDS.
// ---------------------------------------------------------------------------
__global__ __launch_bounds__(128)
void attn_kernel(const unsigned short* __restrict__ qb,
                 const unsigned short* __restrict__ kb,
                 const unsigned short* __restrict__ vb,
                 unsigned short* __restrict__ ob)
{
    constexpr int LK = 72;    // sK  row stride (ushort), 144B
    constexpr int LV = 40;    // sVt row stride
    constexpr int LP = 40;    // sP  row stride
    __shared__ unsigned short sK [32 * LK];    // K tile, (key, d) row-major
    __shared__ unsigned short sVt[64 * LV];    // V tile transposed, (d, key)
    __shared__ unsigned short sP [4][16 * LP]; // per-wave P scratch (16 q x 32 keys)

    const int tid  = threadIdx.x;
    const int lane = tid & 31;
    const int wave = tid >> 5;
    const int ln   = lane & 15;
    const int lg   = lane >> 4;

    const int bh = blockIdx.y;                 // b*H + h
    const int b  = bh >> 4;
    const int h  = bh & 15;
    const int q0 = blockIdx.x * 64 + wave * 16;

    const unsigned short* Q = qb + (size_t)bh * Sc * Dc;
    const unsigned short* K = kb + (size_t)bh * Sc * Dc;
    const unsigned short* V = vb + (size_t)bh * Sc * Dc;

    // Q fragments for d-chunks [0,32) and [32,64); resident all iterations
    FragAB aq[2];
    #pragma unroll
    for (int c = 0; c < 2; ++c) {
        const unsigned short* qr = Q + (size_t)(q0 + ln) * Dc + c * 32;
        aq[c].q[0] = *(const u32x4*)(qr + lg * 8);
        aq[c].q[1] = *(const u32x4*)(qr + 16 + lg * 8);
    }

    v8f o[4];
    #pragma unroll
    for (int t = 0; t < 4; ++t)
        #pragma unroll
        for (int r = 0; r < 8; ++r) o[t][r] = 0.0f;
    float mrow[8], lrow[8];
    #pragma unroll
    for (int j = 0; j < 8; ++j) { mrow[j] = -3.0e38f; lrow[j] = 0.0f; }

    const float scale = 0.125f;                // 1/sqrt(64)

    for (int kb0 = 0; kb0 < Sc; kb0 += 32) {
        // --- stage K tile async: 32 x 64 ushort, 256 16B segments ---
        #pragma unroll
        for (int s = 0; s < 2; ++s) {
            int seg = tid + s * 128;
            int row = seg >> 3;
            int so  = (seg & 7) * 8;
            async_copy16(&K[(size_t)(kb0 + row) * Dc + so], &sK[row * LK + so]);
        }
        // --- prefetch next key-block's K and V (global_prefetch_b8) ---
        if (kb0 + 32 < Sc) {
            __builtin_prefetch(K + (size_t)(kb0 + 32) * Dc + tid * 16, 0, 3);
            __builtin_prefetch(V + (size_t)(kb0 + 32) * Dc + tid * 16, 0, 3);
        }
        // --- stage V transposed: thread reads 16 contiguous d, scatters to sVt ---
        {
            int key = tid >> 2;
            int dq  = (tid & 3) * 16;
            const unsigned short* vr = &V[(size_t)(kb0 + key) * Dc + dq];
            unsigned short tmp[16];
            *(u32x4*)(tmp)     = *(const u32x4*)(vr);
            *(u32x4*)(tmp + 8) = *(const u32x4*)(vr + 8);
            #pragma unroll
            for (int e = 0; e < 16; ++e)
                sVt[(dq + e) * LV + key] = tmp[e];
        }
        async_wait0();
        __syncthreads();

        // --- scores: S = Q @ K^T, two 16x16 tiles over 32 keys ---
        v8f sc[2];
        #pragma unroll
        for (int nb = 0; nb < 2; ++nb) {
            #pragma unroll
            for (int r = 0; r < 8; ++r) sc[nb][r] = 0.0f;
            #pragma unroll
            for (int c = 0; c < 2; ++c) {
                FragAB bk;
                const unsigned short* kr = &sK[(nb * 16 + ln) * LK + c * 32];
                bk.q[0] = *(const u32x4*)(kr + lg * 8);
                bk.q[1] = *(const u32x4*)(kr + 16 + lg * 8);
                sc[nb] = __builtin_amdgcn_wmma_f32_16x16x32_bf16(
                    false, aq[c].v, false, bk.v, (short)0, sc[nb], false, false);
            }
        }

        // --- online softmax (row = j + 8*lg, spread over 16 lanes) ---
        #pragma unroll
        for (int j = 0; j < 8; ++j) {
            float s0 = sc[0][j] * scale;
            float s1 = sc[1][j] * scale;
            float m  = fmaxf(s0, s1);
            #pragma unroll
            for (int off = 1; off < 16; off <<= 1)
                m = fmaxf(m, __shfl_xor(m, off, 16));
            float mn = fmaxf(mrow[j], m);
            float al = __expf(mrow[j] - mn);
            float p0 = __expf(s0 - mn);
            float p1 = __expf(s1 - mn);
            float rs = p0 + p1;
            #pragma unroll
            for (int off = 1; off < 16; off <<= 1)
                rs += __shfl_xor(rs, off, 16);
            lrow[j] = lrow[j] * al + rs;
            mrow[j] = mn;
            #pragma unroll
            for (int t = 0; t < 4; ++t) o[t][j] *= al;
            // store P (bf16) into per-wave LDS scratch, (q row, key) layout
            unsigned short* pr = &sP[wave][(j + 8 * lg) * LP];
            pr[ln]      = f32_to_bf16(p0);
            pr[ln + 16] = f32_to_bf16(p1);
        }

        // --- P as A-fragment (16 q x 32 keys) ---
        FragAB ap;
        {
            const unsigned short* pr = &sP[wave][ln * LP];
            ap.q[0] = *(const u32x4*)(pr + lg * 8);
            ap.q[1] = *(const u32x4*)(pr + 16 + lg * 8);
        }
        // --- O += P @ V : 4 d-tiles of 16 columns ---
        #pragma unroll
        for (int t = 0; t < 4; ++t) {
            FragAB bv;
            const unsigned short* vr = &sVt[(t * 16 + ln) * LV];
            bv.q[0] = *(const u32x4*)(vr + lg * 8);
            bv.q[1] = *(const u32x4*)(vr + 16 + lg * 8);
            o[t] = __builtin_amdgcn_wmma_f32_16x16x32_bf16(
                false, ap.v, false, bv.v, (short)0, o[t], false, false);
        }
        __syncthreads();
    }

    // --- normalize and write O tile into (B,S,E) bf16 buffer ---
    #pragma unroll
    for (int t = 0; t < 4; ++t) {
        #pragma unroll
        for (int j = 0; j < 8; ++j) {
            int s   = q0 + j + 8 * lg;
            int col = h * 64 + t * 16 + ln;
            float val = o[t][j] / lrow[j];
            ob[((size_t)(b * Sc + s)) * Ec + col] = f32_to_bf16(val);
        }
    }
}

// ---------------------------------------------------------------------------
// Host-side launcher
// ---------------------------------------------------------------------------
extern "C" void kernel_launch(void* const* d_in, const int* in_sizes, int n_in,
                              void* d_out, int out_size, void* d_ws, size_t ws_size,
                              hipStream_t stream)
{
    const float* x    = (const float*)d_in[0];   // (B,S,E)
    const float* cosb = (const float*)d_in[1];   // (S,D)
    const float* sinb = (const float*)d_in[2];   // (S,D)
    const float* wqkv = (const float*)d_in[3];   // (3E,E)
    const float* wout = (const float*)d_in[4];   // (E,E)
    float* out = (float*)d_out;                  // (B,S,E) f32

    // workspace carve-up (bf16 buffers); ob aliases xb (xb dead after QKV GEMM)
    char* p = (char*)d_ws;
    unsigned short* xb  = (unsigned short*)p;  p += (size_t)Mtot * Ec * 2;         // 16 MB
    unsigned short* wqb = (unsigned short*)p;  p += (size_t)3 * Ec * Ec * 2;       //  6 MB
    unsigned short* wob = (unsigned short*)p;  p += (size_t)Ec * Ec * 2;           //  2 MB
    unsigned short* qb  = (unsigned short*)p;  p += (size_t)Bc * Hc * Sc * Dc * 2; // 16 MB
    unsigned short* kbuf= (unsigned short*)p;  p += (size_t)Bc * Hc * Sc * Dc * 2; // 16 MB
    unsigned short* vbuf= (unsigned short*)p;  p += (size_t)Bc * Hc * Sc * Dc * 2; // 16 MB
    unsigned short* ob  = xb;                  // reuse

    // 1) f32 -> bf16
    {
        int n1 = Mtot * Ec;          // 8,388,608
        int n2 = 3 * Ec * Ec;        // 3,145,728
        int n3 = Ec * Ec;            // 1,048,576
        cvt_f32_bf16<<<n1 / (256 * 4), 256, 0, stream>>>(x,    xb,  n1);
        cvt_f32_bf16<<<n2 / (256 * 4), 256, 0, stream>>>(wqkv, wqb, n2);
        cvt_f32_bf16<<<n3 / (256 * 4), 256, 0, stream>>>(wout, wob, n3);
    }
    // 2) QKV projection, scatter to (B,H,S,D)
    gemm_bf16<1, 3 * Ec><<<dim3((3 * Ec) / 128, Mtot / 128), 256, 0, stream>>>(
        xb, wqb, nullptr, qb, kbuf, vbuf);
    // 3) RoPE on q and k
    {
        int nthr = Bc * Hc * Sc * 32;    // 4,194,304
        rope_kernel<<<nthr / 256, 256, 0, stream>>>(qb,   cosb, sinb);
        rope_kernel<<<nthr / 256, 256, 0, stream>>>(kbuf, cosb, sinb);
    }
    // 4) attention -> (B,S,E) bf16
    attn_kernel<<<dim3(Sc / 64, Bc * Hc), 128, 0, stream>>>(qb, kbuf, vbuf, ob);
    // 5) output projection -> f32
    gemm_bf16<0, Ec><<<dim3(Ec / 128, Mtot / 128), 256, 0, stream>>>(
        ob, wob, out, nullptr, nullptr, nullptr);
}
